// _STSGCL_52707838656915
// MI455X (gfx1250) — compile-verified, matrix operand order checked by reading, not run on hardware
//
#include <hip/hip_runtime.h>
#include <hip/hip_bf16.h>
#include <stdint.h>

typedef __attribute__((ext_vector_type(16))) __bf16 v16bf;
typedef __attribute__((ext_vector_type(8)))  float  v8f;

union FragBF { v16bf bf; uint4 u[2]; };

#define NN   1024   // nodes
#define CCH  64     // channels
#define TT   12     // time steps
#define BB   16     // batch
#define NWIN 10     // windows
#define W3   3072   // 3*N

#define KSTEP  128              // K-values staged per pipeline stage
#define ROWB   272              // LDS bytes per c-row (256B data + 16B pad)
#define BUFB   (64 * ROWB)      // 17408 B per slab buffer
#define NSTAGE (W3 / KSTEP)     // 24

// ---- CDNA5 async global->LDS copy (VGLOBAL, tracked by ASYNCcnt) ----------
__device__ __forceinline__ void async_b128(unsigned lds_off, unsigned voff,
                                           unsigned long long sbase) {
    asm volatile("global_load_async_to_lds_b128 %0, %1, %2"
                 :: "v"(lds_off), "v"(voff), "s"(sbase) : "memory");
}
template<int N>
__device__ __forceinline__ void wait_async() {
    asm volatile("s_wait_asynccnt %0" :: "i"(N) : "memory");
}

// ---------------------------------------------------------------------------
// Prep 1: xeT[b][t][c][n] = bf16( x[b,t,n,c] + temb[t,c] + semb[n,c] )
// ---------------------------------------------------------------------------
__global__ __launch_bounds__(256)
void k_embed_transpose(const float* __restrict__ x,
                       const float* __restrict__ temb,
                       const float* __restrict__ semb,
                       __bf16* __restrict__ xeT) {
    int strip = blockIdx.x;        // 0 .. 16*12*16-1
    int n0 = (strip & 15) * 64;
    int bt = strip >> 4;           // b*12 + t
    int t  = bt % TT;
    __shared__ __bf16 sT[64][66];  // [c][n], padded
    const float* xp = x + (size_t)bt * NN * CCH;
    const float* tp = temb + t * CCH;
    #pragma unroll
    for (int r = 0; r < 16; ++r) {
        int idx = r * 256 + threadIdx.x;
        int nn = idx >> 6, cc = idx & 63;
        float v = xp[(size_t)(n0 + nn) * CCH + cc] + tp[cc]
                + semb[(size_t)(n0 + nn) * CCH + cc];
        sT[cc][nn] = (__bf16)v;
    }
    __syncthreads();
    __bf16* op = xeT + (size_t)bt * CCH * NN;
    #pragma unroll
    for (int r = 0; r < 16; ++r) {
        int idx = r * 256 + threadIdx.x;
        int cc = idx >> 6, nn = idx & 63;
        op[(size_t)cc * NN + n0 + nn] = sT[cc][nn];
    }
}

// ---------------------------------------------------------------------------
// Prep 2: fp32 -> bf16 conversion (A and W)
// ---------------------------------------------------------------------------
__global__ __launch_bounds__(256)
void k_cvt(const float* __restrict__ in, __bf16* __restrict__ out, size_t n) {
    size_t i = (size_t)blockIdx.x * blockDim.x + threadIdx.x;
    size_t stride = (size_t)gridDim.x * blockDim.x;
    for (; i < n; i += stride) out[i] = (__bf16)in[i];
}

__device__ __forceinline__ void cvt8(const float* __restrict__ p, FragBF& f, int base) {
    #pragma unroll
    for (int e = 0; e < 8; ++e) f.bf[base + e] = (__bf16)p[e];
}

// ---------------------------------------------------------------------------
// Main fused kernel, one launch per GCN iteration j (ITER).
// Grid: (48 vblocks, 16 batch, 10 windows); block = 128 (4 waves of 32).
// Pipeline: async-load B slab (64c x 128k bf16) double-buffered in LDS,
// 16 WMMAs per stage per wave, fused z-GEMM + GLU epilogue.
// ---------------------------------------------------------------------------
template<int ITER>
__global__ __launch_bounds__(128)
void k_gcn(const __bf16* __restrict__ Abf,   // [3072][3072]
           const __bf16* __restrict__ Bsrc,  // ITER0: xeT [B][T][64][1024]; ITER1: h2T [g][b][64][3072]
           const __bf16* __restrict__ Wbf,   // [NW][2][128][64]
           const float*  __restrict__ bias,  // [NW][2][128]
           float*        __restrict__ out,   // [B][NW][1024][64]
           __bf16*       __restrict__ h2T)   // [NW][B][64][3072]
{
    const int g = blockIdx.z, b = blockIdx.y, vblk = blockIdx.x;
    const int wave = threadIdx.x >> 5, lane = threadIdx.x & 31;
    const int lo = lane & 15, hi = lane >> 4;
    const int v0 = vblk * 64 + wave * 16;

    __shared__ __align__(16) unsigned char sBraw[2 * BUFB]; // double-buffered slab
    __shared__ __align__(16) float sY[4][16][68];           // per-wave y transpose

    // LDS byte offset of slab base (generic ptr low 32 bits = LDS address)
    const unsigned sB_base = (unsigned)(uintptr_t)(void*)sBraw;

    // per-thread async chunk geometry: 1024 x 16B chunks, 8 per thread
    const int GROWB = (ITER == 0) ? (NN * 2) : (W3 * 2);    // global row stride (bytes)
    unsigned voff[8], loff[8];
    #pragma unroll
    for (int i = 0; i < 8; ++i) {
        int q = i * 128 + (int)threadIdx.x;
        int c = q >> 4, ch = q & 15;
        voff[i] = (unsigned)(c * GROWB + ch * 16);
        loff[i] = sB_base + (unsigned)(c * ROWB + ch * 16);
    }

    // uniform global byte base of the slab for K-offset k0
    auto src_base = [&](int k0) -> unsigned long long {
        if (ITER == 0) {
            int tp = k0 >> 10, kin = k0 & 1023;
            return (unsigned long long)(uintptr_t)Bsrc +
                   2ull * ((unsigned long long)((b * TT + g + tp) * CCH) * NN + (unsigned)kin);
        } else {
            return (unsigned long long)(uintptr_t)Bsrc +
                   2ull * ((unsigned long long)((g * BB + b) * CCH) * W3 + (unsigned)k0);
        }
    };

    // prologue: stage 0 -> buffer 0
    {
        unsigned long long sb = src_base(0);
        #pragma unroll
        for (int i = 0; i < 8; ++i) async_b128(loff[i], voff[i], sb);
    }

    v8f acc[4] = {};
    const __bf16* Arow = Abf + (size_t)(v0 + lo) * W3 + (hi ? 8 : 0);

    for (int s = 0; s < NSTAGE; ++s) {
        const int k0 = s * KSTEP;
        const unsigned bufoff = (unsigned)((s & 1) * BUFB);

        // prefetch next stage into the other buffer, then wait for this stage
        if (s + 1 < NSTAGE) {
            unsigned long long sb = src_base(k0 + KSTEP);
            const unsigned nbuf = (unsigned)(((s + 1) & 1) * BUFB);
            #pragma unroll
            for (int i = 0; i < 8; ++i) async_b128(loff[i] + nbuf, voff[i], sb);
            wait_async<8>();   // stage s complete (in-order), stage s+1 in flight
        } else {
            wait_async<0>();
        }

        // A fragments for the whole 128-wide stage (4 x 16x32 bf16)
        FragBF fa[4];
        #pragma unroll
        for (int f = 0; f < 4; ++f) {
            const __bf16* ap = Arow + k0 + f * 32;
            fa[f].u[0] = *(const uint4*)(ap);
            fa[f].u[1] = *(const uint4*)(ap + 16);
        }

        __syncthreads();       // slab visible to all waves

        const unsigned char* sbuf = sBraw + bufoff;
        #pragma unroll
        for (int f = 0; f < 4; ++f) {
            #pragma unroll
            for (int t = 0; t < 4; ++t) {
                FragBF fb;
                const unsigned char* bp = sbuf + (t * 16 + lo) * ROWB + f * 64 + (hi ? 16 : 0);
                fb.u[0] = *(const uint4*)(bp);
                fb.u[1] = *(const uint4*)(bp + 32);
                acc[t] = __builtin_amdgcn_wmma_f32_16x16x32_bf16(
                    false, fa[f].bf, false, fb.bf, (short)0, acc[t], false, false);
            }
        }
        __syncthreads();       // all waves done reading before buffer reuse
    }

    // ---- transpose y tile through LDS (acc layout: n=lo, m=i+8*hi) ----
    #pragma unroll
    for (int t = 0; t < 4; ++t)
        #pragma unroll
        for (int i = 0; i < 8; ++i)
            sY[wave][i + 8 * hi][t * 16 + lo] = acc[t][i];
    __syncthreads();

    // ---- z = y @ W^T : K=64 (2 steps of 32), N=128 (8 tiles) ----
    v8f zacc[8] = {};
    const __bf16* wbase = Wbf + (size_t)(g * 2 + ITER) * 128 * 64;
    #pragma unroll
    for (int ks = 0; ks < 2; ++ks) {
        const int kk = ks * 32 + (hi ? 8 : 0);
        FragBF fa;                                // A = y tile, row m = lo
        cvt8(&sY[wave][lo][kk],      fa, 0);
        cvt8(&sY[wave][lo][kk + 16], fa, 8);
        #pragma unroll
        for (int ot = 0; ot < 8; ++ot) {
            FragBF fb;                            // B = W[g,j][o][c], col o, k = c
            const __bf16* wp = wbase + (size_t)(ot * 16 + lo) * 64 + kk;
            fb.u[0] = *(const uint4*)(wp);
            fb.u[1] = *(const uint4*)(wp + 16);
            zacc[ot] = __builtin_amdgcn_wmma_f32_16x16x32_bf16(
                false, fa.bf, false, fb.bf, (short)0, zacc[ot], false, false);
        }
    }

    // ---- bias ----
    const float* bp = bias + (size_t)(g * 2 + ITER) * 128;
    #pragma unroll
    for (int ot = 0; ot < 8; ++ot) {
        float bv = bp[ot * 16 + lo];
        #pragma unroll
        for (int i = 0; i < 8; ++i) zacc[ot][i] += bv;
    }

    // ---- GLU: h' = lhs * sigmoid(rhs); tiles t and t+4 share lane mapping ----
    v8f hnew[4];
    #pragma unroll
    for (int t = 0; t < 4; ++t)
        #pragma unroll
        for (int i = 0; i < 8; ++i) {
            float r = zacc[t + 4][i];
            float s = 1.0f / (1.0f + __expf(-r));
            hnew[t][i] = zacc[t][i] * s;
        }

    // ---- j==0: store h' transposed (bf16) for next iteration ----
    if (ITER == 0) {
        __bf16* hb = h2T + (size_t)(g * BB + b) * CCH * W3;
        #pragma unroll
        for (int t = 0; t < 4; ++t)
            #pragma unroll
            for (int i = 0; i < 8; ++i) {
                int c = t * 16 + lo;
                int w = v0 + i + 8 * hi;
                hb[(size_t)c * W3 + w] = (__bf16)hnew[t][i];
            }
    }

    // ---- middle slab -> output (j==0 write, j==1 max) ----
    if (v0 >= NN && v0 < 2 * NN) {
        float* op = out + ((size_t)(b * NWIN + g) * NN + (v0 - NN)) * CCH;
        #pragma unroll
        for (int t = 0; t < 4; ++t)
            #pragma unroll
            for (int i = 0; i < 8; ++i) {
                int m = i + 8 * hi;
                int c = t * 16 + lo;
                float* p = op + (size_t)m * CCH + c;
                if (ITER == 0) *p = hnew[t][i];
                else           *p = fmaxf(*p, hnew[t][i]);
            }
    }
}

// ---------------------------------------------------------------------------
// Workspace layout (bytes, 256-aligned):
//   xeT bf16 @ 0 (25,165,824) | Abf bf16 @ 25165824 (18,874,368)
//   Wbf bf16 @ 44040192 (327,680) | h2T bf16 @ 44367872 (62,914,560)
// Total: 107,282,432 bytes
// ---------------------------------------------------------------------------
extern "C" void kernel_launch(void* const* d_in, const int* in_sizes, int n_in,
                              void* d_out, int out_size, void* d_ws, size_t ws_size,
                              hipStream_t stream) {
    const float* x    = (const float*)d_in[0];
    const float* A    = (const float*)d_in[1];
    const float* W    = (const float*)d_in[2];
    const float* bias = (const float*)d_in[3];
    const float* temb = (const float*)d_in[4];
    const float* semb = (const float*)d_in[5];
    float* out = (float*)d_out;

    char* ws = (char*)d_ws;
    __bf16* xeT = (__bf16*)(ws);
    __bf16* Abf = (__bf16*)(ws + 25165824);
    __bf16* Wbf = (__bf16*)(ws + 44040192);
    __bf16* h2T = (__bf16*)(ws + 44367872);

    k_embed_transpose<<<dim3(16 * 12 * 16), dim3(256), 0, stream>>>(x, temb, semb, xeT);
    k_cvt<<<dim3(4608), dim3(256), 0, stream>>>(A, Abf, (size_t)W3 * W3);
    k_cvt<<<dim3(160),  dim3(256), 0, stream>>>(W, Wbf, (size_t)NWIN * 2 * 128 * 64);

    dim3 grid(48, BB, NWIN), blk(128);
    k_gcn<0><<<grid, blk, 0, stream>>>(Abf, xeT, Wbf, bias, out, h2T);
    k_gcn<1><<<grid, blk, 0, stream>>>(Abf, h2T, Wbf, bias, out, h2T);
}
